// DreamGraphReasoner_48799418417874
// MI455X (gfx1250) — matrix operand order repaired and checked by reading, not compiled, not run on hardware
//
#include <hip/hip_runtime.h>
#include <math.h>

// ---------------------------------------------------------------------------
// DreamGraphReasoner — MI455X (gfx1250, wave32, WMMA) implementation, v4.
//
//  * All dense GEMMs + attention matmuls: v_wmma_f32_16x16x32_bf16,
//    32x32 tile per wave (2x2 subtiles, 4 WMMAs / K-step).
//  * Main GEMM: double-buffered ASYNC pipeline — global_load_async_to_lds_b128
//    copies 16B/lane straight into the fragment-major LDS slot (swizzle done
//    by per-lane LDS addresses), tracked with ASYNCcnt (s_wait_asynccnt);
//    fragments then read via ds_load_b128 and fed to WMMA. No staging VGPRs.
//  * Weights pre-converted ONCE to bf16 in (N,K) transposed layout; producer
//    kernels emit bf16 activations (v_cvt_pk_bf16_f32).
//  * Edge predictor folded algebraically (no O(N^2 D^2) GEMM, ef never
//    materialized); fused per-edge wave kernel.
//
// Input flattening assumption: depth-first, dict-insertion order (79 leaves).
// ---------------------------------------------------------------------------

typedef __attribute__((ext_vector_type(16))) __bf16 v16bf;
typedef __attribute__((ext_vector_type(2)))  __bf16 v2bf;
typedef __attribute__((ext_vector_type(8)))  float  v8f;
typedef unsigned short ushort_t;

#define NB    8
#define NNODE 128
#define DIM   512
#define NHEAD 8
#define HDIM  64
#define NPAIR (NB * NNODE * NNODE)

// ------------------------------- device utils ------------------------------

__device__ inline unsigned bfpack2(float lo, float hi) {   // 2 x f32 -> packed bf16
  v2bf t; t[0] = (__bf16)lo; t[1] = (__bf16)hi;            // v_cvt_pk_bf16_f32
  union { v2bf v; unsigned u; } cv; cv.v = t; return cv.u;
}
__device__ inline ushort_t bfb(float f) {
  __bf16 h = (__bf16)f;
  ushort_t s; __builtin_memcpy(&s, &h, 2); return s;
}
__device__ inline float wred_sum(float v) {
  #pragma unroll
  for (int s = 16; s > 0; s >>= 1) v += __shfl_xor(v, s, 32);
  return v;
}
__device__ inline float geluf(float x) {  // tanh approximation (jax default)
  float t = tanhf(0.7978845608028654f * (x + 0.044715f * x * x * x));
  return 0.5f * x * (1.0f + t);
}
__device__ inline float sigmf(float x) { return 1.0f / (1.0f + expf(-x)); }

// --------------------- fragment-major LDS staging helpers ------------------
// LDS layout: l[sub][lane][c] (bf16 bits). Fragment (sub,lane) = 32 bytes.
// A element (row r, k): frag lane=(r&15)+16*(k>>4),    c=k&15
// B element (col,  k): frag lane=(col&15)+16*(k>>4),    c=k&15
// 8-bf16 chunks (kk%8==0) stay in one 8-block -> single 16B slot.

__device__ inline int aidx(int r, int kk) {
  return (((r >> 4) << 5) + (r & 15) + ((kk >> 4) << 4)) * 16 + (kk & 15);
}
__device__ inline int bidx(int col, int kk) {
  return (((col >> 4) << 5) + (col & 15) + ((kk >> 4) << 4)) * 16 + (kk & 15);
}
__device__ inline void stage_a8(ushort_t* lA, int r, int kk, uint4 v) {
  *(uint4*)(lA + aidx(r, kk)) = v;
}
__device__ inline void stage_b8(ushort_t* lB, int col, int kk, uint4 v) {
  *(uint4*)(lB + bidx(col, kk)) = v;
}
// 4-float chunk converted to bf16 (only used for the f32 V gather in attv)
__device__ inline void stage_b4f(ushort_t* lB, int col, int kk, float4 v) {
  uint2 pk; pk.x = bfpack2(v.x, v.y); pk.y = bfpack2(v.z, v.w);
  *(uint2*)(lB + (((col >> 4) << 5) + (col & 15) + ((kk >> 4) << 4)) * 16 +
            (kk & 15)) = pk;
}
__device__ inline v16bf ld_frag(const ushort_t* l, int sub, int lane) {
  return *(const v16bf*)(l + ((sub << 5) + lane) * 16);
}
// Low 32 bits of a generic pointer to __shared__ = wave-relative LDS offset
// (ISA 10.2: LDS aperture address is truncated to addr[31:0]).
__device__ inline unsigned lds_off(const ushort_t* p) {
  return (unsigned)(unsigned long long)p;
}
#define WMMA_BF16(A, B, C) \
  __builtin_amdgcn_wmma_f32_16x16x32_bf16(false, (A), false, (B), (short)0, (C), false, false)

// ------------------------------ generic GEMM -------------------------------
// Cf/Cb[M,N] = epi( A[M,K]bf16 @ Bt[N,K]bf16^T + bias (+resid) ).
// One wave per 32x32 tile. EPI: 0=none 1=+bias 2=+bias,gelu 3=+bias,+resid.
// OB: 0=f32 out, 1=bf16 out, 2=both. Requires K%32==0.
template <int EPI, int OB>
__global__ void gemm_k(const ushort_t* __restrict__ A, int lda,
                       const ushort_t* __restrict__ Bt, int ldb,
                       const float* __restrict__ bias,
                       const float* __restrict__ resid, int ldr,
                       float* __restrict__ Cf, ushort_t* __restrict__ Cb,
                       int ldc, int M, int N, int K) {
  __shared__ ushort_t lA[2][2 * 32 * 16] __attribute__((aligned(32)));
  __shared__ ushort_t lB[2][2 * 32 * 16] __attribute__((aligned(32)));
  const int lane = threadIdx.x & 31;
  const int half = lane >> 4, l16 = lane & 15;
  const int tM = blockIdx.y * 32, tN = blockIdx.x * 32;
  v8f a00 = {}, a01 = {}, a10 = {}, a11 = {};

  const bool full = (tM + 32 <= M) && (tN + 32 <= N);
  if (full) {
    // ---- async double-buffered pipeline (ASYNCcnt) ----
    auto issue = [&](int buf, int k0c) {
      #pragma unroll
      for (int it = 0; it < 4; ++it) {
        int q = lane + 32 * it;
        int r = q >> 2, kk = (q & 3) * 8;
        const ushort_t* ga = A + (long)(tM + r) * lda + k0c + kk;
        unsigned la = lds_off(&lA[buf][aidx(r, kk)]);
        asm volatile("global_load_async_to_lds_b128 %0, %1, off"
                     :: "v"(la), "v"((unsigned long long)ga) : "memory");
        const ushort_t* gb = Bt + (long)(tN + r) * ldb + k0c + kk;  // col=r
        unsigned lb = lds_off(&lB[buf][bidx(r, kk)]);
        asm volatile("global_load_async_to_lds_b128 %0, %1, off"
                     :: "v"(lb), "v"((unsigned long long)gb) : "memory");
      }
    };
    const int nsteps = K / 32;
    issue(0, 0);
    for (int s = 0; s < nsteps; ++s) {
      int buf = s & 1;
      if (s + 1 < nsteps) {
        // prior fragment reads of the other buffer must retire before overwrite
        asm volatile("s_wait_dscnt 0x0" ::: "memory");
        issue(buf ^ 1, (s + 1) * 32);
        asm volatile("s_wait_asynccnt 0x8" ::: "memory");  // drain older batch
      } else {
        asm volatile("s_wait_asynccnt 0x0" ::: "memory");
      }
      v16bf fa0 = ld_frag(lA[buf], 0, lane), fa1 = ld_frag(lA[buf], 1, lane);
      v16bf fb0 = ld_frag(lB[buf], 0, lane), fb1 = ld_frag(lB[buf], 1, lane);
      a00 = WMMA_BF16(fa0, fb0, a00);
      a01 = WMMA_BF16(fa0, fb1, a01);
      a10 = WMMA_BF16(fa1, fb0, a10);
      a11 = WMMA_BF16(fa1, fb1, a11);
    }
  } else {
    // ---- bounds-checked synchronous path (edge tiles, e.g. M=8 head) ----
    for (int k0 = 0; k0 < K; k0 += 32) {
      if (k0 + 32 < K)
        __builtin_prefetch(&A[(long)(tM + (lane >> 2)) * lda + k0 + 32], 0, 0);
      #pragma unroll
      for (int it = 0; it < 4; ++it) {
        int q = lane + 32 * it;
        int r = q >> 2, kk = (q & 3) * 8;
        int gr = tM + r;
        uint4 v = {0u, 0u, 0u, 0u};
        if (gr < M) v = *(const uint4*)&A[(long)gr * lda + k0 + kk];
        stage_a8(lA[0], r, kk, v);
      }
      #pragma unroll
      for (int it = 0; it < 4; ++it) {
        int q = lane + 32 * it;
        int col = q >> 2, kk = (q & 3) * 8;
        int gc = tN + col;
        uint4 v = {0u, 0u, 0u, 0u};
        if (gc < N) v = *(const uint4*)&Bt[(long)gc * ldb + k0 + kk];
        stage_b8(lB[0], col, kk, v);
      }
      v16bf fa0 = ld_frag(lA[0], 0, lane), fa1 = ld_frag(lA[0], 1, lane);
      v16bf fb0 = ld_frag(lB[0], 0, lane), fb1 = ld_frag(lB[0], 1, lane);
      a00 = WMMA_BF16(fa0, fb0, a00);
      a01 = WMMA_BF16(fa0, fb1, a01);
      a10 = WMMA_BF16(fa1, fb0, a10);
      a11 = WMMA_BF16(fa1, fb1, a11);
    }
  }

  #pragma unroll
  for (int sa = 0; sa < 2; ++sa) {
    #pragma unroll
    for (int sb = 0; sb < 2; ++sb) {
      v8f acc = (sa == 0) ? (sb == 0 ? a00 : a01) : (sb == 0 ? a10 : a11);
      #pragma unroll
      for (int r = 0; r < 8; ++r) {
        int m = tM + sa * 16 + r + half * 8;
        int n = tN + sb * 16 + l16;
        if (m < M && n < N) {
          float v = acc[r];
          if (EPI != 0) v += bias[n];
          if (EPI == 2) v = geluf(v);
          if (EPI == 3) v += resid[(long)m * ldr + n];
          if (OB != 1) Cf[(long)m * ldc + n] = v;
          if (OB != 0) Cb[(long)m * ldc + n] = bfb(v);
        }
      }
    }
  }
}

// ---------------------- attention scores (WMMA, fused) ---------------------
// grid (4 ktile, 4 qtile, 64 bh), block 32. S[z,i,j] = adj? q.k/8+eb : -inf
__global__ void scores_k(const ushort_t* __restrict__ Q,
                         const ushort_t* __restrict__ Kt,
                         const float* __restrict__ eb, const int* __restrict__ adj,
                         float* __restrict__ S, int layer) {
  __shared__ ushort_t lA[2 * 32 * 16] __attribute__((aligned(32)));
  __shared__ ushort_t lB[2 * 32 * 16] __attribute__((aligned(32)));
  const int lane = threadIdx.x & 31;
  const int half = lane >> 4, l16 = lane & 15;
  const int z = blockIdx.z, b = z >> 3, h = z & 7;
  const int tQ = blockIdx.y * 32, tK = blockIdx.x * 32;
  const ushort_t* qb = Q  + ((long)b * NNODE) * DIM + h * HDIM;
  const ushort_t* kb = Kt + ((long)b * NNODE) * DIM + h * HDIM;
  v8f a00 = {}, a01 = {}, a10 = {}, a11 = {};
  #pragma unroll
  for (int k0 = 0; k0 < HDIM; k0 += 32) {
    #pragma unroll
    for (int it = 0; it < 4; ++it) {
      int q = lane + 32 * it;
      int r = q >> 2, kk = (q & 3) * 8;
      stage_a8(lA, r, kk, *(const uint4*)&qb[(long)(tQ + r) * DIM + k0 + kk]);
    }
    #pragma unroll
    for (int it = 0; it < 4; ++it) {            // K^T: contiguous along k
      int q = lane + 32 * it;
      int col = q >> 2, kk = (q & 3) * 8;
      stage_b8(lB, col, kk, *(const uint4*)&kb[(long)(tK + col) * DIM + k0 + kk]);
    }
    v16bf fa0 = ld_frag(lA, 0, lane), fa1 = ld_frag(lA, 1, lane);
    v16bf fb0 = ld_frag(lB, 0, lane), fb1 = ld_frag(lB, 1, lane);
    a00 = WMMA_BF16(fa0, fb0, a00);
    a01 = WMMA_BF16(fa0, fb1, a01);
    a10 = WMMA_BF16(fa1, fb0, a10);
    a11 = WMMA_BF16(fa1, fb1, a11);
  }
  const float* ebz = eb + ((((long)layer * NB + b) * NHEAD + h) * (NNODE * NNODE));
  #pragma unroll
  for (int sa = 0; sa < 2; ++sa) {
    #pragma unroll
    for (int sb = 0; sb < 2; ++sb) {
      v8f acc = (sa == 0) ? (sb == 0 ? a00 : a01) : (sb == 0 ? a10 : a11);
      #pragma unroll
      for (int r = 0; r < 8; ++r) {
        int i = tQ + sa * 16 + r + half * 8;
        int j = tK + sb * 16 + l16;
        float v = adj[i * NNODE + j] ? (acc[r] * 0.125f + ebz[i * NNODE + j])
                                     : -INFINITY;
        S[((long)z * NNODE + i) * NNODE + j] = v;
      }
    }
  }
}

// --------------------------- att = probs @ V (WMMA) ------------------------
// grid (2 ntile, 4 qtile, 64 bh), block 32. probs bf16, V f32 (converted here)
__global__ void attv_k(const ushort_t* __restrict__ Pr, const float* __restrict__ V,
                       ushort_t* __restrict__ attb) {
  __shared__ ushort_t lA[2 * 32 * 16] __attribute__((aligned(32)));
  __shared__ ushort_t lB[2 * 32 * 16] __attribute__((aligned(32)));
  const int lane = threadIdx.x & 31;
  const int half = lane >> 4, l16 = lane & 15;
  const int z = blockIdx.z, b = z >> 3, h = z & 7;
  const int tQ = blockIdx.y * 32, tN = blockIdx.x * 32;
  const ushort_t* pr = Pr + (long)z * NNODE * NNODE;
  const float* vb = V + ((long)b * NNODE) * DIM + h * HDIM;
  v8f a00 = {}, a01 = {}, a10 = {}, a11 = {};
  #pragma unroll
  for (int k0 = 0; k0 < NNODE; k0 += 32) {
    #pragma unroll
    for (int it = 0; it < 4; ++it) {
      int q = lane + 32 * it;
      int r = q >> 2, kk = (q & 3) * 8;
      stage_a8(lA, r, kk, *(const uint4*)&pr[(long)(tQ + r) * NNODE + k0 + kk]);
    }
    #pragma unroll
    for (int it = 0; it < 8; ++it) {            // V: gather 4 k-rows per col
      int col = lane, kk = it * 4;
      float4 v;
      v.x = vb[(long)(k0 + kk + 0) * DIM + tN + col];
      v.y = vb[(long)(k0 + kk + 1) * DIM + tN + col];
      v.z = vb[(long)(k0 + kk + 2) * DIM + tN + col];
      v.w = vb[(long)(k0 + kk + 3) * DIM + tN + col];
      stage_b4f(lB, col, kk, v);
    }
    v16bf fa0 = ld_frag(lA, 0, lane), fa1 = ld_frag(lA, 1, lane);
    v16bf fb0 = ld_frag(lB, 0, lane), fb1 = ld_frag(lB, 1, lane);
    a00 = WMMA_BF16(fa0, fb0, a00);
    a01 = WMMA_BF16(fa0, fb1, a01);
    a10 = WMMA_BF16(fa1, fb0, a10);
    a11 = WMMA_BF16(fa1, fb1, a11);
  }
  #pragma unroll
  for (int sa = 0; sa < 2; ++sa) {
    #pragma unroll
    for (int sb = 0; sb < 2; ++sb) {
      v8f acc = (sa == 0) ? (sb == 0 ? a00 : a01) : (sb == 0 ? a10 : a11);
      #pragma unroll
      for (int r = 0; r < 8; ++r) {
        int qi = tQ + sa * 16 + r + half * 8;
        int n  = tN + sb * 16 + l16;
        attb[((long)b * NNODE + qi) * DIM + h * HDIM + n] = bfb(acc[r]);
      }
    }
  }
}

// ---------------- softmax over key axis; writes bf16 probs -----------------
__global__ void softmax_k(const float* __restrict__ S, ushort_t* __restrict__ Pb,
                          int rows) {
  int row = blockIdx.x * (blockDim.x >> 5) + (threadIdx.x >> 5);
  if (row >= rows) return;
  int lane = threadIdx.x & 31;
  const float* p = S + (long)row * NNODE;
  float4 v = *(const float4*)&p[lane * 4];
  float m = fmaxf(fmaxf(v.x, v.y), fmaxf(v.z, v.w));
  #pragma unroll
  for (int s = 16; s > 0; s >>= 1) m = fmaxf(m, __shfl_xor(m, s, 32));
  v.x = expf(v.x - m); v.y = expf(v.y - m);
  v.z = expf(v.z - m); v.w = expf(v.w - m);
  float inv = 1.0f / wred_sum(v.x + v.y + v.z + v.w);
  uint2 pk;
  pk.x = bfpack2(v.x * inv, v.y * inv);
  pk.y = bfpack2(v.z * inv, v.w * inv);
  *(uint2*)&Pb[(long)row * NNODE + lane * 4] = pk;
}

// ------------------- LayerNorm rows (f32 in -> bf16 out) -------------------
__global__ void ln_k(const float* __restrict__ X, const float* __restrict__ g,
                     const float* __restrict__ b, ushort_t* __restrict__ Yb,
                     int rows) {
  int row = blockIdx.x * (blockDim.x >> 5) + (threadIdx.x >> 5);
  if (row >= rows) return;
  int lane = threadIdx.x & 31;
  const float* x = X + (long)row * DIM;
  ushort_t* y = Yb + (long)row * DIM;
  float4 tv[4];
  float s = 0.f;
  #pragma unroll
  for (int t = 0; t < 4; ++t) {
    tv[t] = *(const float4*)&x[lane * 4 + 128 * t];
    s += tv[t].x + tv[t].y + tv[t].z + tv[t].w;
  }
  float mean = wred_sum(s) * (1.0f / DIM);
  float s2 = 0.f;
  #pragma unroll
  for (int t = 0; t < 4; ++t) {
    float cx = tv[t].x - mean, cy = tv[t].y - mean;
    float cz = tv[t].z - mean, cw = tv[t].w - mean;
    s2 += cx * cx + cy * cy + cz * cz + cw * cw;
  }
  float rs = rsqrtf(wred_sum(s2) * (1.0f / DIM) + 1e-5f);
  #pragma unroll
  for (int t = 0; t < 4; ++t) {
    int d = lane * 4 + 128 * t;
    float4 g4 = *(const float4*)&g[d];
    float4 b4 = *(const float4*)&b[d];
    uint2 pk;
    pk.x = bfpack2((tv[t].x - mean) * rs * g4.x + b4.x,
                   (tv[t].y - mean) * rs * g4.y + b4.y);
    pk.y = bfpack2((tv[t].z - mean) * rs * g4.z + b4.z,
                   (tv[t].w - mean) * rs * g4.w + b4.w);
    *(uint2*)&y[d] = pk;
  }
}

// -------------------------- conversion kernels -----------------------------
__global__ void cvt_k(const float* __restrict__ s, ushort_t* __restrict__ d,
                      long n4) {
  long idx = (long)blockIdx.x * blockDim.x + threadIdx.x;
  if (idx >= n4) return;
  float4 v = *(const float4*)&s[idx * 4];
  uint2 pk; pk.x = bfpack2(v.x, v.y); pk.y = bfpack2(v.z, v.w);
  *(uint2*)&d[idx * 4] = pk;
}
// W (K,N) f32 row-major -> Wt (N,K) bf16
__global__ void cvtT_k(const float* __restrict__ W, ushort_t* __restrict__ Wt,
                       int K, int N) {
  long idx = (long)blockIdx.x * blockDim.x + threadIdx.x;
  long tot = (long)N * (K / 4);
  if (idx >= tot) return;
  int n = (int)(idx / (K / 4));
  int kc = (int)(idx % (K / 4)) * 4;
  float a = W[(long)(kc + 0) * N + n];
  float b = W[(long)(kc + 1) * N + n];
  float c = W[(long)(kc + 2) * N + n];
  float d = W[(long)(kc + 3) * N + n];
  uint2 pk; pk.x = bfpack2(a, b); pk.y = bfpack2(c, d);
  *(uint2*)&Wt[(long)n * K + kc] = pk;
}

// ----------------------- edge-predictor folded tables ----------------------
struct FoldArgs {
  const float* g[3];
  const float* bb[3];
  const float* W[3];
  const float* be[3];
};

__global__ void fold_k(FoldArgs fa, const float* __restrict__ E,
                       float* __restrict__ Wp, float* __restrict__ cp,
                       float* __restrict__ Gt, float* __restrict__ Sv,
                       float* __restrict__ mu, float* __restrict__ M4) {
  int tid = threadIdx.x;
  for (int t = tid; t < DIM * 24; t += blockDim.x) {
    int d = t / 24, c = t % 24, l = c >> 3, h = c & 7;
    Wp[t] = fa.g[l][d] * fa.W[l][d * 8 + h];
  }
  __syncthreads();
  if (tid < 24) {
    int c = tid, l = c >> 3, h = c & 7;
    float s = 0.f;
    for (int d = 0; d < DIM; ++d) s += fa.bb[l][d] * fa.W[l][d * 8 + h];
    cp[c] = s + fa.be[l][h];
  } else if (tid < 120) {
    int idx = tid - 24, t4 = idx / 24, c = idx % 24;
    float s = 0.f;
    for (int d = 0; d < DIM; ++d) s += E[t4 * DIM + d] * Wp[d * 24 + c];
    Gt[t4 * 24 + c] = s;
  } else if (tid < 144) {
    int c = tid - 120;
    float s = 0.f;
    for (int d = 0; d < DIM; ++d) s += Wp[d * 24 + c];
    Sv[c] = s;
  } else if (tid < 148) {
    int t4 = tid - 144;
    float s = 0.f;
    for (int d = 0; d < DIM; ++d) s += E[t4 * DIM + d];
    mu[t4] = s * (1.0f / DIM);
  } else if (tid < 164) {
    int idx = tid - 148, i = idx >> 2, j = idx & 3;
    float s = 0.f;
    for (int d = 0; d < DIM; ++d) s += E[i * DIM + d] * E[j * DIM + d];
    M4[i * 4 + j] = s;
  }
}

// ------------------------ fused edge pipeline (per edge) -------------------
__global__ void edge_pipeline_k(
    const float* __restrict__ P, const float* __restrict__ Qb,
    const int* __restrict__ adj,
    const float* __restrict__ epg, const float* __restrict__ epb,
    const float* __restrict__ W2, const float* __restrict__ b2,
    const float* __restrict__ mu, const float* __restrict__ M4,
    const float* __restrict__ Gt, const float* __restrict__ Sv,
    const float* __restrict__ cp, float* __restrict__ eb) {
  int wave = blockIdx.x * (blockDim.x >> 5) + (threadIdx.x >> 5);
  if (wave >= NPAIR) return;
  int lane = threadIdx.x & 31;
  int b = wave >> 14;
  int ij = wave & 16383;
  int i = ij >> 7, j = ij & 127;
  if (adj[i * NNODE + j] == 0) return;

  const float* pi = P  + ((long)(b * NNODE + i)) * DIM;
  const float* qj = Qb + ((long)(b * NNODE + j)) * DIM;
  float4 tv[4];
  float s = 0.f;
  #pragma unroll
  for (int t = 0; t < 4; ++t) {
    int d = lane * 4 + 128 * t;
    float4 p4 = *(const float4*)&pi[d];
    float4 q4 = *(const float4*)&qj[d];
    tv[t].x = p4.x + q4.x; tv[t].y = p4.y + q4.y;
    tv[t].z = p4.z + q4.z; tv[t].w = p4.w + q4.w;
    s += tv[t].x + tv[t].y + tv[t].z + tv[t].w;
  }
  float mean = wred_sum(s) * (1.0f / DIM);
  float s2 = 0.f;
  #pragma unroll
  for (int t = 0; t < 4; ++t) {
    float cx = tv[t].x - mean, cy = tv[t].y - mean;
    float cz = tv[t].z - mean, cw = tv[t].w - mean;
    s2 += cx * cx + cy * cy + cz * cz + cw * cw;
  }
  float rs = rsqrtf(wred_sum(s2) * (1.0f / DIM) + 1e-5f);
  float a0 = 0, a1 = 0, a2 = 0, a3 = 0;
  #pragma unroll
  for (int t = 0; t < 4; ++t) {
    int d0 = lane * 4 + 128 * t;
    float4 g4 = *(const float4*)&epg[d0];
    float4 b4 = *(const float4*)&epb[d0];
    float xs[4] = {tv[t].x, tv[t].y, tv[t].z, tv[t].w};
    float gs[4] = {g4.x, g4.y, g4.z, g4.w};
    float bs[4] = {b4.x, b4.y, b4.z, b4.w};
    #pragma unroll
    for (int u = 0; u < 4; ++u) {
      float h = (xs[u] - mean) * rs * gs[u] + bs[u];
      float g = geluf(h);
      float4 wr = *(const float4*)&W2[(long)(d0 + u) * 4];
      a0 += g * wr.x; a1 += g * wr.y; a2 += g * wr.z; a3 += g * wr.w;
    }
  }
  a0 = wred_sum(a0) + b2[0]; a1 = wred_sum(a1) + b2[1];
  a2 = wred_sum(a2) + b2[2]; a3 = wred_sum(a3) + b2[3];
  float mx = fmaxf(fmaxf(a0, a1), fmaxf(a2, a3));
  float e0 = expf(a0 - mx), e1 = expf(a1 - mx), e2 = expf(a2 - mx), e3 = expf(a3 - mx);
  float inv = 1.0f / (e0 + e1 + e2 + e3);
  float pv[4] = { e0 * inv, e1 * inv, e2 * inv, e3 * inv };
  float m_ef = pv[0] * mu[0] + pv[1] * mu[1] + pv[2] * mu[2] + pv[3] * mu[3];
  float msq = 0.f;
  #pragma unroll
  for (int a = 0; a < 4; ++a)
    #pragma unroll
    for (int c = 0; c < 4; ++c) msq += pv[a] * pv[c] * M4[a * 4 + c];
  msq *= (1.0f / DIM);
  float rs2 = rsqrtf(msq - m_ef * m_ef + 1e-5f);
  if (lane < 24) {
    int c = lane;
    float acc = pv[0] * Gt[c] + pv[1] * Gt[24 + c] + pv[2] * Gt[48 + c] + pv[3] * Gt[72 + c];
    float val = rs2 * (acc - m_ef * Sv[c]) + cp[c];
    int l = c >> 3, h = c & 7;
    eb[((((long)l * NB + b) * NHEAD + h) * (NNODE * NNODE)) + i * NNODE + j] = val;
  }
}

// ------------------------------- GRU update --------------------------------
__global__ void gru_k(const float* __restrict__ gi, const float* __restrict__ gh,
                      float* __restrict__ mem, ushort_t* __restrict__ memb,
                      float* __restrict__ x) {
  long idx = (long)blockIdx.x * blockDim.x + threadIdx.x;   // float4 index
  if (idx >= (long)NB * NNODE * DIM / 4) return;
  long row = idx >> 7;
  int d = (int)(idx & 127) * 4;
  const float* gir = gi + row * (3 * DIM);
  const float* ghr = gh + row * (3 * DIM);
  float4 ir = *(const float4*)&gir[d];
  float4 iz = *(const float4*)&gir[DIM + d];
  float4 in = *(const float4*)&gir[2 * DIM + d];
  float4 hr = *(const float4*)&ghr[d];
  float4 hz = *(const float4*)&ghr[DIM + d];
  float4 hn = *(const float4*)&ghr[2 * DIM + d];
  float4 mo = *(const float4*)&mem[row * DIM + d];
  float4 out;
  { float r = sigmf(ir.x + hr.x), z = sigmf(iz.x + hz.x);
    float n = tanhf(in.x + r * hn.x); out.x = (1.f - z) * n + z * mo.x; }
  { float r = sigmf(ir.y + hr.y), z = sigmf(iz.y + hz.y);
    float n = tanhf(in.y + r * hn.y); out.y = (1.f - z) * n + z * mo.y; }
  { float r = sigmf(ir.z + hr.z), z = sigmf(iz.z + hz.z);
    float n = tanhf(in.z + r * hn.z); out.z = (1.f - z) * n + z * mo.z; }
  { float r = sigmf(ir.w + hr.w), z = sigmf(iz.w + hz.w);
    float n = tanhf(in.w + r * hn.w); out.w = (1.f - z) * n + z * mo.w; }
  *(float4*)&mem[row * DIM + d] = out;
  *(float4*)&x[row * DIM + d] = out;
  uint2 pk; pk.x = bfpack2(out.x, out.y); pk.y = bfpack2(out.z, out.w);
  *(uint2*)&memb[row * DIM + d] = pk;
}

__global__ void zero_k(float* __restrict__ p, long n4) {
  long idx = (long)blockIdx.x * blockDim.x + threadIdx.x;
  if (idx < n4) *(float4*)&p[idx * 4] = make_float4(0.f, 0.f, 0.f, 0.f);
}
__global__ void zero_us_k(ushort_t* __restrict__ p, long n8) {
  long idx = (long)blockIdx.x * blockDim.x + threadIdx.x;
  if (idx < n8) *(uint4*)&p[idx * 8] = make_uint4(0u, 0u, 0u, 0u);
}

// ------------- mean-pool over nodes + LayerNorm (bf16 out) -----------------
__global__ void pool_k(const float* __restrict__ x, const float* __restrict__ g,
                       const float* __restrict__ b, ushort_t* __restrict__ gpb) {
  __shared__ float sm[256];
  int bi = blockIdx.x, tid = threadIdx.x;
  float v[2];
  #pragma unroll
  for (int t = 0; t < 2; ++t) {
    int d = tid + 256 * t;
    float s = 0.f;
    for (int n = 0; n < NNODE; ++n) s += x[((long)bi * NNODE + n) * DIM + d];
    v[t] = s * (1.0f / NNODE);
  }
  sm[tid] = v[0] + v[1];
  __syncthreads();
  for (int s = 128; s > 0; s >>= 1) { if (tid < s) sm[tid] += sm[tid + s]; __syncthreads(); }
  float mean = sm[0] * (1.0f / DIM);
  __syncthreads();
  float c0 = v[0] - mean, c1 = v[1] - mean;
  sm[tid] = c0 * c0 + c1 * c1;
  __syncthreads();
  for (int s = 128; s > 0; s >>= 1) { if (tid < s) sm[tid] += sm[tid + s]; __syncthreads(); }
  float rs = rsqrtf(sm[0] * (1.0f / DIM) + 1e-5f);
  #pragma unroll
  for (int t = 0; t < 2; ++t) {
    int d = tid + 256 * t;
    gpb[(long)bi * DIM + d] = bfb((v[t] - mean) * rs * g[d] + b[d]);
  }
}

// ---------------------------------- host -----------------------------------

static void launch_gemm(int epi, int ob, const ushort_t* A, int lda,
                        const ushort_t* Bt, int ldb, const float* bias,
                        const float* resid, int ldr, float* Cf, ushort_t* Cb,
                        int ldc, int M, int N, int K, hipStream_t s) {
  dim3 g((N + 31) / 32, (M + 31) / 32, 1);
  dim3 blk(32, 1, 1);
  if (epi == 0)
    gemm_k<0, 0><<<g, blk, 0, s>>>(A, lda, Bt, ldb, bias, resid, ldr, Cf, Cb, ldc, M, N, K);
  else if (epi == 1 && ob == 0)
    gemm_k<1, 0><<<g, blk, 0, s>>>(A, lda, Bt, ldb, bias, resid, ldr, Cf, Cb, ldc, M, N, K);
  else if (epi == 1 && ob == 1)
    gemm_k<1, 1><<<g, blk, 0, s>>>(A, lda, Bt, ldb, bias, resid, ldr, Cf, Cb, ldc, M, N, K);
  else if (epi == 2)
    gemm_k<2, 1><<<g, blk, 0, s>>>(A, lda, Bt, ldb, bias, resid, ldr, Cf, Cb, ldc, M, N, K);
  else if (epi == 3 && ob == 0)
    gemm_k<3, 0><<<g, blk, 0, s>>>(A, lda, Bt, ldb, bias, resid, ldr, Cf, Cb, ldc, M, N, K);
  else
    gemm_k<3, 2><<<g, blk, 0, s>>>(A, lda, Bt, ldb, bias, resid, ldr, Cf, Cb, ldc, M, N, K);
}

extern "C" void kernel_launch(void* const* d_in, const int* in_sizes, int n_in,
                              void* d_out, int out_size, void* d_ws, size_t ws_size,
                              hipStream_t stream) {
  (void)in_sizes; (void)n_in; (void)out_size; (void)ws_size;
  auto F = [&](int i) { return (const float*)d_in[i]; };
  int p = 0;
  const float* node_emb = F(p++);
  const float* epl1_w = F(p++); const float* epl1_b = F(p++);
  const float* epln_g = F(p++); const float* epln_b = F(p++);
  const float* epl2_w = F(p++); const float* epl2_b = F(p++);
  const float* edgeE  = F(p++);
  struct Layer {
    const float *nn_g, *nn_b, *en_g, *en_b, *fn_g, *fn_b;
    const float *qw, *qb, *kw, *kb, *vw, *vb, *ow, *ob;
    const float *ebw, *ebb, *f1w, *f1b, *f2w, *f2b;
  } L[3];
  for (int l = 0; l < 3; ++l) {
    L[l].nn_g = F(p++); L[l].nn_b = F(p++);
    L[l].en_g = F(p++); L[l].en_b = F(p++);
    L[l].fn_g = F(p++); L[l].fn_b = F(p++);
    L[l].qw = F(p++); L[l].qb = F(p++);
    L[l].kw = F(p++); L[l].kb = F(p++);
    L[l].vw = F(p++); L[l].vb = F(p++);
    L[l].ow = F(p++); L[l].ob = F(p++);
    L[l].ebw = F(p++); L[l].ebb = F(p++);
    L[l].f1w = F(p++); L[l].f1b = F(p++);
    L[l].f2w = F(p++); L[l].f2b = F(p++);
  }
  const float* wih = F(p++); const float* whh = F(p++);
  const float* bih = F(p++); const float* bhh = F(p++);
  const float* fln_g = F(p++); const float* fln_b = F(p++);
  const float* fl1_w = F(p++); const float* fl1_b = F(p++);
  const float* fl2_w = F(p++); const float* fl2_b = F(p++);
  const int*   adj   = (const int*)d_in[p++];
  float* outp = (float*)d_out;

  // ---- workspace carve-up: f32 region then bf16 (ushort) region ----
  float* wsf = (float*)d_ws;
  size_t fo = 0;
  auto af = [&](size_t n) { size_t r = fo; fo += (n + 63) & ~(size_t)63; return r; };
  const size_t TOK = (size_t)NB * NNODE * DIM;
  size_t P_  = af(TOK),      Qb_ = af(TOK);
  size_t Wp_ = af(DIM * 24), cp_ = af(24), Gt_ = af(96);
  size_t Sv_ = af(24),       mu_ = af(4),  M4_ = af(16);
  size_t eb_ = af((size_t)3 * NB * NHEAD * NNODE * NNODE);
  size_t x_  = af(TOK), v_ = af(TOK);
  size_t sc_ = af((size_t)NB * NHEAD * NNODE * NNODE);
  size_t gi_ = af((size_t)NB * NNODE * 3 * DIM);
  size_t gh_ = af((size_t)NB * NNODE * 3 * DIM);
  size_t mem_ = af(TOK);

  size_t FBYTES = (fo * sizeof(float) + 255) & ~(size_t)255;
  ushort_t* wsu = (ushort_t*)((char*)d_ws + FBYTES);
  size_t uo = 0;
  auto au = [&](size_t n) { size_t r = uo; uo += (n + 63) & ~(size_t)63; return r; };
  const size_t WSQ = (size_t)DIM * DIM;                // 512x512
  size_t nodeb_ = au(TOK);
  size_t e1a_ = au(WSQ), e1b_ = au(WSQ);
  size_t qwt_[3], kwt_[3], vwt_[3], owt_[3], f1wt_[3], f2wt_[3];
  for (int l = 0; l < 3; ++l) {
    qwt_[l] = au(WSQ); kwt_[l] = au(WSQ); vwt_[l] = au(WSQ); owt_[l] = au(WSQ);
    f1wt_[l] = au(4 * WSQ); f2wt_[l] = au(4 * WSQ);
  }
  size_t wihb_ = au(3 * WSQ), whhb_ = au(3 * WSQ);
  size_t fl1t_ = au(2 * WSQ), fl2t_ = au(2 * WSQ);
  size_t nxb_ = au(TOK), qb_ = au(TOK), kb_ = au(TOK);
  size_t attb_ = au(TOK), xb_ = au(TOK), memb_ = au(TOK);
  size_t ffb_ = au(4 * TOK);
  size_t prb_ = au((size_t)NB * NHEAD * NNODE * NNODE);
  size_t gpb_ = au(NB * DIM), f1b_ = au(NB * 2 * DIM);

  auto cvt = [&](const float* s, ushort_t* d, long n) {
    cvt_k<<<(unsigned)((n / 4 + 255) / 256), 256, 0, stream>>>(s, d, n / 4);
  };
  auto cvtT = [&](const float* W, ushort_t* Wt, int K, int N) {
    long tot = (long)N * (K / 4);
    cvtT_k<<<(unsigned)((tot + 255) / 256), 256, 0, stream>>>(W, Wt, K, N);
  };

  // ---- one-time (per call) conversions ----
  cvt(node_emb, wsu + nodeb_, TOK);
  cvtT(epl1_w, wsu + e1a_, DIM, DIM);
  cvtT(epl1_w + (size_t)DIM * DIM, wsu + e1b_, DIM, DIM);
  for (int l = 0; l < 3; ++l) {
    cvtT(L[l].qw, wsu + qwt_[l], DIM, DIM);
    cvtT(L[l].kw, wsu + kwt_[l], DIM, DIM);
    cvtT(L[l].vw, wsu + vwt_[l], DIM, DIM);
    cvtT(L[l].ow, wsu + owt_[l], DIM, DIM);
    cvtT(L[l].f1w, wsu + f1wt_[l], DIM, 4 * DIM);     // (512,2048)->(2048,512)
    cvtT(L[l].f2w, wsu + f2wt_[l], 4 * DIM, DIM);     // (2048,512)->(512,2048)
  }
  cvt(wih, wsu + wihb_, 3 * WSQ);                      // already (out,in)
  cvt(whh, wsu + whhb_, 3 * WSQ);
  cvtT(fl1_w, wsu + fl1t_, DIM, 2 * DIM);
  cvtT(fl2_w, wsu + fl2t_, 2 * DIM, DIM);

  // ---- init: x = node_embeddings, mem = 0 ----
  hipMemcpyAsync(wsf + x_, node_emb, TOK * sizeof(float),
                 hipMemcpyDeviceToDevice, stream);
  zero_k<<<(TOK / 4 + 255) / 256, 256, 0, stream>>>(wsf + mem_, (long)(TOK / 4));
  zero_us_k<<<(TOK / 8 + 255) / 256, 256, 0, stream>>>(wsu + memb_, (long)(TOK / 8));

  // ---- fold edge-bias tables ----
  FoldArgs fa;
  for (int l = 0; l < 3; ++l) {
    fa.g[l] = L[l].en_g; fa.bb[l] = L[l].en_b;
    fa.W[l] = L[l].ebw;  fa.be[l] = L[l].ebb;
  }
  fold_k<<<1, 256, 0, stream>>>(fa, edgeE, wsf + Wp_, wsf + cp_, wsf + Gt_,
                                wsf + Sv_, wsf + mu_, wsf + M4_);

  // ---- edge predictor projections: P = x@W1a + b1 ; Qb = x@W1b ----
  launch_gemm(1, 0, wsu + nodeb_, DIM, wsu + e1a_, DIM, epl1_b, nullptr, 0,
              wsf + P_, nullptr, DIM, NB * NNODE, DIM, DIM, stream);
  launch_gemm(0, 0, wsu + nodeb_, DIM, wsu + e1b_, DIM, nullptr, nullptr, 0,
              wsf + Qb_, nullptr, DIM, NB * NNODE, DIM, DIM, stream);

  // ---- fused edge pipeline -> eb for all 3 layers (edges only) ----
  edge_pipeline_k<<<NPAIR / 8, 256, 0, stream>>>(
      wsf + P_, wsf + Qb_, adj, epln_g, epln_b, epl2_w, epl2_b,
      wsf + mu_, wsf + M4_, wsf + Gt_, wsf + Sv_, wsf + cp_, wsf + eb_);

  const int ROWS = NB * NNODE;  // 1024
  for (int l = 0; l < 3; ++l) {
    ln_k<<<ROWS / 8, 256, 0, stream>>>(wsf + x_, L[l].nn_g, L[l].nn_b,
                                       wsu + nxb_, ROWS);
    launch_gemm(1, 1, wsu + nxb_, DIM, wsu + qwt_[l], DIM, L[l].qb, nullptr, 0,
                nullptr, wsu + qb_, DIM, ROWS, DIM, DIM, stream);
    launch_gemm(1, 1, wsu + nxb_, DIM, wsu + kwt_[l], DIM, L[l].kb, nullptr, 0,
                nullptr, wsu + kb_, DIM, ROWS, DIM, DIM, stream);
    launch_gemm(1, 0, wsu + nxb_, DIM, wsu + vwt_[l], DIM, L[l].vb, nullptr, 0,
                wsf + v_, nullptr, DIM, ROWS, DIM, DIM, stream);
    scores_k<<<dim3(4, 4, 64), 32, 0, stream>>>(wsu + qb_, wsu + kb_, wsf + eb_,
                                                adj, wsf + sc_, l);
    softmax_k<<<(NB * NHEAD * NNODE) / 8, 256, 0, stream>>>(
        wsf + sc_, wsu + prb_, NB * NHEAD * NNODE);
    attv_k<<<dim3(2, 4, 64), 32, 0, stream>>>(wsu + prb_, wsf + v_, wsu + attb_);
    launch_gemm(3, 0, wsu + attb_, DIM, wsu + owt_[l], DIM, L[l].ob, wsf + x_,
                DIM, wsf + x_, nullptr, DIM, ROWS, DIM, DIM, stream);
    ln_k<<<ROWS / 8, 256, 0, stream>>>(wsf + x_, L[l].fn_g, L[l].fn_b,
                                       wsu + nxb_, ROWS);
    launch_gemm(2, 1, wsu + nxb_, DIM, wsu + f1wt_[l], DIM, L[l].f1b, nullptr, 0,
                nullptr, wsu + ffb_, 4 * DIM, ROWS, 4 * DIM, DIM, stream);
    launch_gemm(3, 2, wsu + ffb_, 4 * DIM, wsu + f2wt_[l], 4 * DIM, L[l].f2b,
                wsf + x_, DIM, wsf + x_, wsu + xb_, DIM, ROWS, DIM, 4 * DIM,
                stream);
    launch_gemm(1, 0, wsu + xb_, DIM, wsu + wihb_, DIM, bih, nullptr, 0,
                wsf + gi_, nullptr, 3 * DIM, ROWS, 3 * DIM, DIM, stream);
    launch_gemm(1, 0, wsu + memb_, DIM, wsu + whhb_, DIM, bhh, nullptr, 0,
                wsf + gh_, nullptr, 3 * DIM, ROWS, 3 * DIM, DIM, stream);
    gru_k<<<(TOK / 4 + 255) / 256, 256, 0, stream>>>(
        wsf + gi_, wsf + gh_, wsf + mem_, wsu + memb_, wsf + x_);
  }

  pool_k<<<NB, 256, 0, stream>>>(wsf + x_, fln_g, fln_b, wsu + gpb_);
  launch_gemm(2, 1, wsu + gpb_, DIM, wsu + fl1t_, DIM, fl1_b, nullptr, 0,
              nullptr, wsu + f1b_, 2 * DIM, NB, 2 * DIM, DIM, stream);
  launch_gemm(1, 0, wsu + f1b_, 2 * DIM, wsu + fl2t_, 2 * DIM, fl2_b, nullptr, 0,
              outp, nullptr, DIM, NB, DIM, 2 * DIM, stream);
}
